// LatentQuantizer_84954453115641
// MI455X (gfx1250) — matrix operand order repaired and checked by reading, not compiled, not run on hardware
//
#include <hip/hip_runtime.h>
#include <hip/hip_bf16.h>
#include <stdint.h>

typedef __attribute__((ext_vector_type(16))) __bf16 v16bf;
typedef __attribute__((ext_vector_type(8)))  float  v8f;

#define N_ROWS 262144
#define DIM    64
#define NEMB   1024
#define NWAVES (N_ROWS / 16)          // 16384
#define NBLKS  (NWAVES / 8)           // 2048

// workspace layout (bytes)
#define WS_P    0                     // packed B (hi+lo interleaved): 256 KB
#define WS_WSQ  262144                // ||w_e||^2 : 4 KB
#define WS_PART 266240                // per-wave loss partials: 64 KB

// d_out layout (floats): zq[N*64] | scaler[N] | redshift[N] | loss[1] | ids[N]
#define OFF_SC   ((size_t)N_ROWS * DIM)
#define OFF_RS   (OFF_SC + N_ROWS)
#define OFF_LOSS (OFF_RS + N_ROWS)
#define OFF_IDS  (OFF_LOSS + 1)

static __device__ __forceinline__ v8f wmma_bf16(v16bf a, v16bf b, v8f c) {
    return __builtin_amdgcn_wmma_f32_16x16x32_bf16(
        /*neg_a=*/false, a, /*neg_b=*/false, b,
        /*c_mod=*/(short)0, c, /*reuse_a=*/false, /*reuse_b=*/false);
}

// ---------------------------------------------------------------------------
// Prep: pack codebook W[D][E] into the per-lane WMMA B layout as split bf16
// (hi, lo), and compute ||w_e||^2.  Packed element order:
//   P[tile][kc][h][lane][i]   (bf16), h=0 hi, h=1 lo
// with B mapping: N = 16*tile + (lane&15), K = 32*kc + 16*(lane>>4) + i.
// One thread per (tile,kc,lane,i): 65536 threads.
// ---------------------------------------------------------------------------
__global__ void __launch_bounds__(256) vq_prep(const float* __restrict__ W,
                                               void* __restrict__ ws) {
    __bf16* P   = (__bf16*)((char*)ws + WS_P);
    float*  wsq = (float*)((char*)ws + WS_WSQ);

    int g = blockIdx.x * 256 + threadIdx.x;      // 0..65535
    int i = g & 15;
    int L = (g >> 4) & 31;
    int c = (g >> 9) & 1;
    int t = g >> 10;                             // 0..63

    int k = 32 * c + 16 * (L >> 4) + i;
    int n = 16 * t + (L & 15);
    float x = W[k * NEMB + n];
    __bf16 hi = (__bf16)x;
    __bf16 lo = (__bf16)(x - (float)hi);
    int base = (((t * 2 + c) * 2 + 0) * 32 + L) * 16 + i;
    P[base]       = hi;
    P[base + 512] = lo;                          // h=1 is +32*16 elements

    if (g < NEMB) {
        float s = 0.f;
        #pragma unroll
        for (int kk = 0; kk < DIM; ++kk) {
            float w = W[kk * NEMB + g];
            s += w * w;
        }
        wsq[g] = s;
    }
}

// ---------------------------------------------------------------------------
// Main: each wave quantizes 16 rows. Block of 8 waves shares LDS-staged
// packed codebook tiles. s = z.W via split-bf16 WMMA (hi*hi + hi*lo + lo*hi),
// distance surrogate = ||w||^2 - 2 s  (row term ||z||^2 is argmin-invariant).
// Staging is software-pipelined: group g+1's global loads are issued before
// group g's 96-WMMA compute, so L2 latency hides behind matrix work.
// ---------------------------------------------------------------------------
__global__ void __launch_bounds__(256) vq_main(const float* __restrict__ Z,
                                               const float* __restrict__ SC,
                                               const float* __restrict__ RS,
                                               const float* __restrict__ W,
                                               float* __restrict__ out,
                                               void* __restrict__ ws) {
    __shared__ __align__(32) uint4 smemRaw[4096];     // 64 KB
    v16bf* smemB = (v16bf*)smemRaw;

    const uint4* Pg  = (const uint4*)((const char*)ws + WS_P);
    const float* wsq = (const float*)((const char*)ws + WS_WSQ);
    float*       prt = (float*)((char*)ws + WS_PART);

    const int tid  = threadIdx.x;
    const int lane = tid & 31;
    const int wave = tid >> 5;
    const int gwave = blockIdx.x * 8 + wave;
    const int r0 = gwave * 16;
    const int halfsel = lane >> 4;      // 0: K-low half, 1: K-high half
    const int mrow    = lane & 15;      // A-matrix row / B,C column for this lane

    // --- build A (16 rows x 64 dims) as split bf16, WMMA A layout ---
    // lane holds row (lane&15); chunk c: K=32c+16*(lane>>4)+i
    const float* zr = Z + (size_t)(r0 + mrow) * DIM;
    v16bf Ahi[2], Alo[2];
    #pragma unroll
    for (int c = 0; c < 2; ++c) {
        int kA = 32 * c + 16 * halfsel;
        const float4* p0 = (const float4*)(zr + kA);
        const float4* p1 = (const float4*)(zr + kA + 8);
        float4 f0 = p0[0], f1 = p0[1], f2 = p1[0], f3 = p1[1];
        float xs[16] = { f0.x, f0.y, f0.z, f0.w, f1.x, f1.y, f1.z, f1.w,
                         f2.x, f2.y, f2.z, f2.w, f3.x, f3.y, f3.z, f3.w };
        #pragma unroll
        for (int i = 0; i < 16; ++i) {
            __bf16 h = (__bf16)xs[i];
            Ahi[c][i] = h;
            Alo[c][i] = (__bf16)(xs[i] - (float)h);
        }
    }

    float best[8];
    int   bid[8];
    #pragma unroll
    for (int j = 0; j < 8; ++j) { best[j] = 3.4e38f; bid[j] = 0; }

    // prologue: issue group 0's staging loads
    uint4 Rbuf[16];
    #pragma unroll
    for (int r = 0; r < 16; ++r)
        Rbuf[r] = Pg[tid + r * 256];

    for (int g = 0; g < 4; ++g) {
        __syncthreads();                          // previous group's readers done
        #pragma unroll
        for (int r = 0; r < 16; ++r)              // commit staged regs to LDS
            smemRaw[tid + r * 256] = Rbuf[r];
        __syncthreads();

        if (g < 3) {                              // prefetch next group (no wait:
            const uint4* src = Pg + (size_t)(g + 1) * 4096;   // consumed next iter)
            #pragma unroll
            for (int r = 0; r < 16; ++r)
                Rbuf[r] = src[tid + r * 256];
        }

        #pragma unroll
        for (int tl = 0; tl < 16; ++tl) {
            const int t = g * 16 + tl;
            v16bf bh0 = smemB[((tl * 2 + 0) * 2 + 0) * 32 + lane];
            v16bf bl0 = smemB[((tl * 2 + 0) * 2 + 1) * 32 + lane];
            v16bf bh1 = smemB[((tl * 2 + 1) * 2 + 0) * 32 + lane];
            v16bf bl1 = smemB[((tl * 2 + 1) * 2 + 1) * 32 + lane];

            v8f acc = {0.f, 0.f, 0.f, 0.f, 0.f, 0.f, 0.f, 0.f};
            acc = wmma_bf16(Ahi[0], bh0, acc);   // hi*hi, K=0..31
            acc = wmma_bf16(Ahi[1], bh1, acc);   // hi*hi, K=32..63
            acc = wmma_bf16(Ahi[0], bl0, acc);   // hi*lo
            acc = wmma_bf16(Ahi[1], bl1, acc);
            acc = wmma_bf16(Alo[0], bh0, acc);   // lo*hi
            acc = wmma_bf16(Alo[1], bh1, acc);

            const int e = t * 16 + mrow;         // this lane's column id
            const float q = wsq[e];
            #pragma unroll
            for (int j = 0; j < 8; ++j) {        // acc[j]: row j + 8*halfsel
                float v = q - 2.0f * acc[j];
                if (v < best[j]) { best[j] = v; bid[j] = e; }
            }
        }
    }

    // min-reduce within each 16-lane half (each half owns 8 rows),
    // tie-break on smaller index to match argmin semantics
    #pragma unroll
    for (int j = 0; j < 8; ++j) {
        #pragma unroll
        for (int off = 8; off >= 1; off >>= 1) {
            float bo = __shfl_xor(best[j], off, 16);
            int   io = __shfl_xor(bid[j],  off, 16);
            if (bo < best[j] || (bo == best[j] && io < bid[j])) {
                best[j] = bo; bid[j] = io;
            }
        }
    }

    // --- outputs: z_q gather, ids, loss partial, pass-throughs ---
    float* zq     = out;
    float* scOut  = out + OFF_SC;
    float* rsOut  = out + OFF_RS;
    float* idsOut = out + OFF_IDS;

    float lsum = 0.f;
    #pragma unroll
    for (int m = 0; m < 16; ++m) {
        // rows 0..7 live in lanes 0-15 (take lane 0); rows 8..15 in lane 16
        int id = __shfl(bid[m & 7], (m < 8) ? 0 : 16, 32);
        const int   row  = r0 + m;
        const float* zrow = Z + (size_t)row * DIM;
        float w0 = W[(size_t)lane        * NEMB + id];
        float w1 = W[(size_t)(lane + 32) * NEMB + id];
        float z0 = zrow[lane];
        float z1 = zrow[lane + 32];
        float* o = zq + (size_t)row * DIM;
        o[lane]      = w0;
        o[lane + 32] = w1;
        float d0 = w0 - z0, d1 = w1 - z1;
        lsum += d0 * d0 + d1 * d1;
        if (lane == 0) idsOut[row] = (float)id;
    }
    if (lane < 16) {
        scOut[r0 + lane] = SC[r0 + lane];
        rsOut[r0 + lane] = RS[r0 + lane];
    }
    #pragma unroll
    for (int off = 16; off >= 1; off >>= 1)
        lsum += __shfl_xor(lsum, off, 32);
    if (lane == 0) prt[gwave] = lsum;
}

// ---------------------------------------------------------------------------
// Deterministic fixed-order loss reduction: 16384 partials -> scalar
// loss = (1 + BETA) * mean((z_q - z)^2)
// ---------------------------------------------------------------------------
__global__ void __launch_bounds__(256) vq_loss(const void* __restrict__ ws,
                                               float* __restrict__ out) {
    const float* prt = (const float*)((const char*)ws + WS_PART);
    __shared__ float red[256];
    float s = 0.f;
    for (int i = threadIdx.x; i < NWAVES; i += 256) s += prt[i];
    red[threadIdx.x] = s;
    __syncthreads();
    for (int st = 128; st > 0; st >>= 1) {
        if (threadIdx.x < st) red[threadIdx.x] += red[threadIdx.x + st];
        __syncthreads();
    }
    if (threadIdx.x == 0) {
        float mse = red[0] / (float)((size_t)N_ROWS * DIM);
        out[OFF_LOSS] = 1.25f * mse;   // (1 + BETA) * mse, sg() is identity
    }
}

extern "C" void kernel_launch(void* const* d_in, const int* in_sizes, int n_in,
                              void* d_out, int out_size, void* d_ws, size_t ws_size,
                              hipStream_t stream) {
    const float* Z  = (const float*)d_in[0];   // latents  [N,64]
    const float* SC = (const float*)d_in[1];   // scaler   [N]
    const float* RS = (const float*)d_in[2];   // redshift [N]
    const float* W  = (const float*)d_in[3];   // codebook [64,1024]
    float* out = (float*)d_out;

    vq_prep<<<256, 256, 0, stream>>>(W, d_ws);
    vq_main<<<NBLKS, 256, 0, stream>>>(Z, SC, RS, W, out, d_ws);
    vq_loss<<<1, 256, 0, stream>>>(d_ws, out);
}
